// RoIHeads_78597901517028
// MI455X (gfx1250) — matrix operand order repaired and checked by reading, not compiled, not run on hardware
//
#include <hip/hip_runtime.h>

typedef __attribute__((ext_vector_type(16))) __bf16 v16bf;
typedef __attribute__((ext_vector_type(8)))  float  v8f;
typedef __attribute__((ext_vector_type(8)))  unsigned int v8u;

#define NUM_CLS   91
#define NC1       90
#define NPROP     2048
#define NIMG      4
#define NCAND     (NPROP * NC1)      /* 184320 candidates per image */
#define DETS      100
#define HID       1024
#define FIN       12544
#define MTOT      (NIMG * NPROP)     /* 8192 */

#define TM 128
#define TN 128
#define TK 32
#define LDP (TK + 2)

__device__ __forceinline__ unsigned short f2bf(float f) {
  __bf16 h = (__bf16)f;                       // RNE, lowers to v_cvt
  return __builtin_bit_cast(unsigned short, h);
}
__device__ __forceinline__ unsigned int pack2bf(float lo, float hi) {
  return (unsigned int)f2bf(lo) | ((unsigned int)f2bf(hi) << 16);
}

// ---------------------------------------------------------------------------
// Tiled GEMM:  C[M,N] = act(A[M,K] @ W[K,N] + bias)
// Double-buffered LDS + register-staged software pipeline: next tile's global
// loads overlap WMMA on the current tile; one barrier per K-step.
// Staging packs bf16 pairs -> ds_store_b32 (conflict-free: stride 17 dwords).
// Grid: x = N-tiles (fastest) so blocks sharing an A panel are co-resident
// and A-panel re-reads hit L2 instead of HBM.
// ---------------------------------------------------------------------------
template <int A_IS_BF16, int OUT_IS_BF16>
__global__ __launch_bounds__(256) void gemm_wmma_bf16(
    const void* __restrict__ Av, const float* __restrict__ W,
    const float* __restrict__ bias, void* __restrict__ Cv,
    int M, int K, int N, int relu)
{
  __shared__ unsigned short sA[2][TM][LDP];   // [buf][m][k] bf16
  __shared__ unsigned short sB[2][TN][LDP];   // [buf][n][k] bf16 (transposed)
  const int tid  = threadIdx.x;
  const int wave = tid >> 5;
  const int lane = tid & 31;
  const int half = lane >> 4;
  const int l16  = lane & 15;
  const int bm = blockIdx.y * TM;
  const int bn = blockIdx.x * TN;
  const int wm = (wave >> 1) * 32;     // 4 waves along M
  const int wn = (wave & 1) * 64;      // 2 waves along N

  v8f acc[2][4] = {};

  // staging (pair-packed): A pair j: row = (tid>>4)+16j, colpair = (tid&15)*2
  //                        B pair j: n = tid&127, kpair = ((tid>>7)+2j)*2
  const int ar0 = tid >> 4, ac0 = (tid & 15) * 2;
  const int bk0 = tid >> 7, bn0 = tid & 127;
  unsigned int aP[8], bP[8];

  auto loadTiles = [&](int k0) {
    if (A_IS_BF16) {
      const unsigned short* A = (const unsigned short*)Av;
      #pragma unroll
      for (int j = 0; j < 8; ++j)
        aP[j] = *(const unsigned int*)(A + (size_t)(bm + ar0 + 16 * j) * K + (k0 + ac0));
    } else {
      const float* A = (const float*)Av;
      #pragma unroll
      for (int j = 0; j < 8; ++j) {
        const float2 f = *(const float2*)(A + (size_t)(bm + ar0 + 16 * j) * K + (k0 + ac0));
        aP[j] = pack2bf(f.x, f.y);
      }
    }
    const int gn = bn + bn0;
    #pragma unroll
    for (int j = 0; j < 8; ++j) {
      const int k = k0 + 2 * (bk0 + 2 * j);
      float lo = 0.0f, hi = 0.0f;
      if (gn < N) {
        lo = W[(size_t)k * N + gn];
        hi = W[(size_t)(k + 1) * N + gn];
      }
      bP[j] = pack2bf(lo, hi);
    }
  };
  auto storeTiles = [&](int buf) {
    #pragma unroll
    for (int j = 0; j < 8; ++j)
      *(unsigned int*)&sA[buf][ar0 + 16 * j][ac0] = aP[j];
    #pragma unroll
    for (int j = 0; j < 8; ++j)
      *(unsigned int*)&sB[buf][bn0][2 * (bk0 + 2 * j)] = bP[j];
  };
  auto compute = [&](int buf) {
    v16bf af[2], bfc[4];
    #pragma unroll
    for (int mi = 0; mi < 2; ++mi) {
      const int arow = wm + mi * 16 + l16;
      v8u u;
      #pragma unroll
      for (int d = 0; d < 8; ++d) {
        // 16-bit A layout: element e -> k = (e/8)*16 + half*8 + (e%8)
        const int kb = ((d >> 2) << 4) + (half << 3) + ((2 * d) & 7);
        u[d] = *(const unsigned int*)&sA[buf][arow][kb];
      }
      af[mi] = __builtin_bit_cast(v16bf, u);
    }
    #pragma unroll
    for (int ni = 0; ni < 4; ++ni) {
      const int bcol = wn + ni * 16 + l16;
      v8u u;
      #pragma unroll
      for (int d = 0; d < 8; ++d)
        u[d] = *(const unsigned int*)&sB[buf][bcol][((d >> 2) << 4) + (half << 3) + ((2 * d) & 7)];
      bfc[ni] = __builtin_bit_cast(v16bf, u);
    }
    #pragma unroll
    for (int mi = 0; mi < 2; ++mi)
      #pragma unroll
      for (int ni = 0; ni < 4; ++ni)
        acc[mi][ni] = __builtin_amdgcn_wmma_f32_16x16x32_bf16(
            false, af[mi], false, bfc[ni], (short)0, acc[mi][ni], false, false);
  };

  // software pipeline over K (double-buffered, one barrier per step)
  loadTiles(0);
  storeTiles(0);
  __syncthreads();
  int cur = 0;
  for (int k0 = TK; k0 < K; k0 += TK) {
    loadTiles(k0);          // global loads in flight
    compute(cur);           // WMMA on current LDS buffer
    storeTiles(cur ^ 1);    // waits loads, fills other buffer
    __syncthreads();        // s_wait_dscnt 0 + barrier
    cur ^= 1;
  }
  compute(cur);

  // epilogue: bias + optional ReLU
  #pragma unroll
  for (int mi = 0; mi < 2; ++mi) {
    #pragma unroll
    for (int ni = 0; ni < 4; ++ni) {
      const int col = bn + wn + ni * 16 + l16;
      if (col < N) {
        const float bs = bias[col];
        #pragma unroll
        for (int v = 0; v < 8; ++v) {
          const int row = bm + wm + mi * 16 + v + half * 8;  // C layout: m = vgpr + half*8
          float x = acc[mi][ni][v] + bs;
          if (relu) x = fmaxf(x, 0.0f);
          if (OUT_IS_BF16)
            ((unsigned short*)Cv)[(size_t)row * N + col] = f2bf(x);
          else
            ((float*)Cv)[(size_t)row * N + col] = x;
        }
      }
    }
  }
}

// ---------------------------------------------------------------------------
// Per-proposal: softmax, decode, clip, validity -> candidate arrays
// ---------------------------------------------------------------------------
__global__ __launch_bounds__(256) void prepare_kernel(
    const float* __restrict__ logits, const float* __restrict__ regr,
    const float* __restrict__ props,
    float* __restrict__ obx, float* __restrict__ sc, float* __restrict__ ar)
{
  const int g = blockIdx.x * blockDim.x + threadIdx.x;
  if (g >= MTOT) return;
  const int b = g / NPROP, n = g % NPROP;
  const float* lg = logits + (size_t)g * NUM_CLS;
  float mx = lg[0];
  for (int c = 1; c < NUM_CLS; ++c) mx = fmaxf(mx, lg[c]);
  float sum = 0.0f;
  for (int c = 0; c < NUM_CLS; ++c) sum += __expf(lg[c] - mx);
  const float inv = 1.0f / sum;

  const float* pr = props + (size_t)g * 4;
  const float w  = pr[2] - pr[0], h  = pr[3] - pr[1];
  const float cx = pr[0] + 0.5f * w, cy = pr[1] + 0.5f * h;
  const float* rg = regr + (size_t)g * (4 * NUM_CLS);
  const float BCLIP = 4.135166556742356f;          // log(1000/16)
  const size_t base = (size_t)b * NCAND + (size_t)n * NC1;

  for (int c = 1; c < NUM_CLS; ++c) {
    float dx = rg[4 * c + 0] * 0.1f;
    float dy = rg[4 * c + 1] * 0.1f;
    float dw = fminf(rg[4 * c + 2] * 0.2f, BCLIP);
    float dh = fminf(rg[4 * c + 3] * 0.2f, BCLIP);
    float pcx = dx * w + cx, pcy = dy * h + cy;
    float pw = __expf(dw) * w, ph = __expf(dh) * h;
    float x1 = fminf(fmaxf(pcx - 0.5f * pw, 0.0f), 800.0f);
    float y1 = fminf(fmaxf(pcy - 0.5f * ph, 0.0f), 800.0f);
    float x2 = fminf(fmaxf(pcx + 0.5f * pw, 0.0f), 800.0f);
    float y2 = fminf(fmaxf(pcy + 0.5f * ph, 0.0f), 800.0f);
    float s  = __expf(lg[c] - mx) * inv;
    bool valid = (s > 0.05f) && ((x2 - x1) >= 0.01f) && ((y2 - y1) >= 0.01f);
    const float off = (float)c * 801.0f;           // per-class NMS offset
    const size_t idx = base + (size_t)(c - 1);
    obx[idx * 4 + 0] = x1 + off;
    obx[idx * 4 + 1] = y1 + off;
    obx[idx * 4 + 2] = x2 + off;
    obx[idx * 4 + 3] = y2 + off;
    ar[idx] = (x2 - x1) * (y2 - y1);
    sc[idx] = valid ? s : -__builtin_inff();
  }
}

// ---------------------------------------------------------------------------
// Deterministic, order-preserving compaction of valid candidates per image.
// Wave ballot/popc prefix + cross-wave LDS scan: original order kept, so
// first-index argmax tie-break over the compacted list matches jnp.argmax.
// ---------------------------------------------------------------------------
__global__ __launch_bounds__(1024) void compact_kernel(
    const float* __restrict__ obx, const float* __restrict__ sc,
    const float* __restrict__ ar,
    float* __restrict__ cob, float* __restrict__ csc,
    float* __restrict__ car, int* __restrict__ cidx, int* __restrict__ nvalid)
{
  const int b = blockIdx.x, t = threadIdx.x;
  const int lane = t & 31, wave = t >> 5;
  const float* s    = sc  + (size_t)b * NCAND;
  const float* ob   = obx + (size_t)b * NCAND * 4;
  const float* area = ar  + (size_t)b * NCAND;
  float* co = cob  + (size_t)b * NCAND * 4;
  float* cs = csc  + (size_t)b * NCAND;
  float* ca = car  + (size_t)b * NCAND;
  int*   ci = cidx + (size_t)b * NCAND;

  __shared__ int wtot[32];
  __shared__ int s_total;
  if (t == 0) s_total = 0;
  __syncthreads();

  for (int base = 0; base < NCAND; base += 1024) {
    const int i = base + t;
    const float v = s[i];                       // NCAND % 1024 == 0
    const int flag = (v > 0.0f) ? 1 : 0;
    const unsigned int m = (unsigned int)__ballot(flag);
    const int lanePre = __popc(m & ((1u << lane) - 1u));
    if (lane == 0) wtot[wave] = __popc(m);
    __syncthreads();
    int waveBase = 0;
    for (int wv = 0; wv < wave; ++wv) waveBase += wtot[wv];
    const int pos = s_total + waveBase + lanePre;
    if (flag) {
      co[(size_t)pos * 4 + 0] = ob[(size_t)i * 4 + 0];
      co[(size_t)pos * 4 + 1] = ob[(size_t)i * 4 + 1];
      co[(size_t)pos * 4 + 2] = ob[(size_t)i * 4 + 2];
      co[(size_t)pos * 4 + 3] = ob[(size_t)i * 4 + 3];
      cs[pos] = v;
      ca[pos] = area[i];
      ci[pos] = i;
    }
    __syncthreads();
    if (t == 0) {
      int ct = 0;
      for (int wv = 0; wv < 32; ++wv) ct += wtot[wv];
      s_total += ct;
    }
    __syncthreads();
  }
  if (t == 0) nvalid[b] = s_total;
}

// ---------------------------------------------------------------------------
// Sequential NMS over compacted candidates: one block per image,
// 100 rounds of (argmax with first-index tie-break) + IoU suppression.
// ---------------------------------------------------------------------------
__global__ __launch_bounds__(1024) void nms_kernel(
    const float* __restrict__ cob, float* __restrict__ csc,
    const float* __restrict__ car, const int* __restrict__ cidx,
    const int* __restrict__ nvalid, float* __restrict__ out)
{
  const int b = blockIdx.x;
  const int t = threadIdx.x;
  const int nv = nvalid[b];
  const float* ob   = cob  + (size_t)b * NCAND * 4;
  float*       s    = csc  + (size_t)b * NCAND;
  const float* area = car  + (size_t)b * NCAND;
  const int*   ci   = cidx + (size_t)b * NCAND;
  __shared__ float rv[1024];
  __shared__ int   ri[1024];
  __shared__ float bb[4];
  __shared__ float bsc, bar;
  __shared__ int bok;

  for (int it = 0; it < DETS; ++it) {
    float best = -__builtin_inff();
    int bi = 0x7fffffff;
    for (int i = t; i < nv; i += 1024) {
      float v = s[i];
      if (v > best || (v == best && i < bi)) { best = v; bi = i; }
    }
    rv[t] = best; ri[t] = bi;
    __syncthreads();
    for (int stp = 512; stp > 0; stp >>= 1) {
      if (t < stp) {
        if (rv[t + stp] > rv[t] || (rv[t + stp] == rv[t] && ri[t + stp] < ri[t])) {
          rv[t] = rv[t + stp]; ri[t] = ri[t + stp];
        }
      }
      __syncthreads();
    }
    if (t == 0) {
      const int bidx = ri[0];
      bsc = rv[0];
      const bool ok = (bidx < nv) && (bsc > 0.0f);
      bok = ok ? 1 : 0;
      int label = 0;
      if (bidx < nv) {
        bb[0] = ob[(size_t)bidx * 4 + 0];
        bb[1] = ob[(size_t)bidx * 4 + 1];
        bb[2] = ob[(size_t)bidx * 4 + 2];
        bb[3] = ob[(size_t)bidx * 4 + 3];
        bar = area[bidx];
        label = ci[bidx] % NC1 + 1;
      } else {
        bb[0] = bb[1] = bb[2] = bb[3] = 0.0f;
        bar = 0.0f;
      }
      const float off = (float)label * 801.0f;
      #pragma unroll
      for (int j = 0; j < 4; ++j)
        out[((size_t)b * DETS + it) * 4 + j] = ok ? (bb[j] - off) : 0.0f;
      out[NIMG * DETS * 4 + b * DETS + it] = ok ? bsc : 0.0f;
      ((int*)out)[NIMG * DETS * 5 + b * DETS + it] = ok ? label : 0;
    }
    __syncthreads();
    if (bok) {
      const float bx1 = bb[0], by1 = bb[1], bx2 = bb[2], by2 = bb[3];
      const float ba = bar;
      for (int i = t; i < nv; i += 1024) {
        float x1 = ob[(size_t)i * 4 + 0], y1 = ob[(size_t)i * 4 + 1];
        float x2 = ob[(size_t)i * 4 + 2], y2 = ob[(size_t)i * 4 + 3];
        float iw = fminf(x2, bx2) - fmaxf(x1, bx1);
        float ih = fminf(y2, by2) - fmaxf(y1, by1);
        float inter = fmaxf(iw, 0.0f) * fmaxf(ih, 0.0f);
        float uni = fmaxf(area[i] + ba - inter, 1e-6f);
        if (inter > 0.5f * uni) s[i] = -__builtin_inff();   // suppresses best itself too
      }
    }
    __syncthreads();
  }
}

// ---------------------------------------------------------------------------
extern "C" void kernel_launch(void* const* d_in, const int* in_sizes, int n_in,
                              void* d_out, int out_size, void* d_ws, size_t ws_size,
                              hipStream_t stream)
{
  (void)in_sizes; (void)n_in; (void)out_size; (void)ws_size;
  const float* box_features = (const float*)d_in[0];
  const float* proposals    = (const float*)d_in[1];
  const float* fc6_w  = (const float*)d_in[2];
  const float* fc6_b  = (const float*)d_in[3];
  const float* fc7_w  = (const float*)d_in[4];
  const float* fc7_b  = (const float*)d_in[5];
  const float* cls_w  = (const float*)d_in[6];
  const float* cls_b  = (const float*)d_in[7];
  const float* bbox_w = (const float*)d_in[8];
  const float* bbox_b = (const float*)d_in[9];

  char* w = (char*)d_ws;
  unsigned short* h1 = (unsigned short*)w; w += (size_t)MTOT * HID * 2;      // bf16 activations
  unsigned short* h2 = (unsigned short*)w; w += (size_t)MTOT * HID * 2;
  float* logits = (float*)w; w += (size_t)MTOT * NUM_CLS * 4;
  float* regr   = (float*)w; w += (size_t)MTOT * 4 * NUM_CLS * 4;
  float* obx    = (float*)w; w += (size_t)NIMG * NCAND * 4 * 4;
  float* scoreb = (float*)w; w += (size_t)NIMG * NCAND * 4;
  float* areab  = (float*)w; w += (size_t)NIMG * NCAND * 4;
  float* cob    = (float*)w; w += (size_t)NIMG * NCAND * 4 * 4;
  float* csc    = (float*)w; w += (size_t)NIMG * NCAND * 4;
  float* car    = (float*)w; w += (size_t)NIMG * NCAND * 4;
  int*   cidx   = (int*)w;   w += (size_t)NIMG * NCAND * 4;
  int*   nvalid = (int*)w;   w += 256;

  const dim3 blk(256);
  // grid.x = N-tiles (fastest) -> A-panel reuse through L2
  gemm_wmma_bf16<0, 1><<<dim3(HID / TN, MTOT / TM), blk, 0, stream>>>(
      box_features, fc6_w, fc6_b, h1, MTOT, FIN, HID, 1);
  gemm_wmma_bf16<1, 1><<<dim3(HID / TN, MTOT / TM), blk, 0, stream>>>(
      h1, fc7_w, fc7_b, h2, MTOT, HID, HID, 1);
  gemm_wmma_bf16<1, 0><<<dim3((NUM_CLS + TN - 1) / TN, MTOT / TM), blk, 0, stream>>>(
      h2, cls_w, cls_b, logits, MTOT, HID, NUM_CLS, 0);
  gemm_wmma_bf16<1, 0><<<dim3((4 * NUM_CLS + TN - 1) / TN, MTOT / TM), blk, 0, stream>>>(
      h2, bbox_w, bbox_b, regr, MTOT, HID, 4 * NUM_CLS, 0);

  prepare_kernel<<<dim3(MTOT / 256), dim3(256), 0, stream>>>(
      logits, regr, proposals, obx, scoreb, areab);
  compact_kernel<<<dim3(NIMG), dim3(1024), 0, stream>>>(
      obx, scoreb, areab, cob, csc, car, cidx, nvalid);
  nms_kernel<<<dim3(NIMG), dim3(1024), 0, stream>>>(
      cob, csc, car, cidx, nvalid, (float*)d_out);
}
